// NERNetCRF_59038620451312
// MI455X (gfx1250) — matrix operand order repaired and checked by reading, not compiled, not run on hardware
//
#include <hip/hip_runtime.h>

// ---------------------------------------------------------------------------
// NER BiLSTM-CRF for MI455X (gfx1250).
//  - bf16 WMMA (v_wmma_f32_16x16x32_bf16) for the fused [x;h]·[Wih;Whh]^T GEMM
//  - ONE persistent kernel per direction: 512 recurrent steps in-kernel with a
//    device-wide split barrier (the recurrence is launch-latency bound, not
//    FLOP bound: 201 MFLOP/step << 1-2us of per-node dispatch overhead)
//  - per-step pointwise LSTM cell + FC->emissions fused in the same kernel
//  - scalar CRF forward (loss) + Viterbi decode (K=9)
// ---------------------------------------------------------------------------

typedef __bf16 bf16_t;
typedef __attribute__((ext_vector_type(16))) __bf16 v16bf;
typedef __attribute__((ext_vector_type(8)))  float  v8f;

#define B_DIM   64
#define T_DIM   512
#define E_DIM   256
#define H_DIM   512
#define FOURH   2048        // 4*H
#define KC      768         // E + H (fused K)
#define K_TAGS  9
#define NBLK    64          // persistent-kernel grid (co-resident by design)

// ---------------------------------------------------------------------------
// Device-wide split barrier: atomic arrive + generation spin through L2.
// Must be called from uniform control flow by all NBLK blocks.
// ---------------------------------------------------------------------------
__device__ __forceinline__ void grid_sync(unsigned* cnt, unsigned* gen)
{
    __syncthreads();
    if (threadIdx.x == 0) {
        __threadfence();                                   // publish phase writes
        unsigned g = __atomic_load_n(gen, __ATOMIC_ACQUIRE);
        if (atomicAdd(cnt, 1u) == NBLK - 1u) {
            __atomic_store_n(cnt, 0u, __ATOMIC_RELAXED);   // reset before release
            __atomic_store_n(gen, g + 1u, __ATOMIC_RELEASE);
        } else {
            while (__atomic_load_n(gen, __ATOMIC_ACQUIRE) == g) {
                __builtin_amdgcn_s_sleep(2);
            }
        }
    }
    __syncthreads();
}

__global__ void barrier_init_kernel(unsigned* __restrict__ bar)
{
    if (threadIdx.x < 4) bar[threadIdx.x] = 0u;            // {cnt,gen} x 2 dirs
}

// ---------------------------------------------------------------------------
// Embedding gather, f32 -> bf16.   xe[b,t,e] = embed[x[b,t], e]
// ---------------------------------------------------------------------------
__global__ void embed_bf16_kernel(const int* __restrict__ x,
                                  const float* __restrict__ embed,
                                  bf16_t* __restrict__ xe, int total)
{
    int i = blockIdx.x * blockDim.x + threadIdx.x;
    if (i >= total) return;
    int e  = i & (E_DIM - 1);
    int bt = i >> 8;                     // E = 256
    xe[i] = (bf16_t)embed[(size_t)x[bt] * E_DIM + e];
}

// ---------------------------------------------------------------------------
// Build concatenated weight wcat[n, k] (n in [0,4H), k in [0,E+H)), bf16.
// ---------------------------------------------------------------------------
__global__ void wcat_kernel(const float* __restrict__ w_ih,
                            const float* __restrict__ w_hh,
                            bf16_t* __restrict__ wcat, int total)
{
    int i = blockIdx.x * blockDim.x + threadIdx.x;
    if (i >= total) return;
    int k = i % KC;
    int n = i / KC;
    float v = (k < E_DIM) ? w_ih[(size_t)n * E_DIM + k]
                          : w_hh[(size_t)n * H_DIM + (k - E_DIM)];
    wcat[i] = (bf16_t)v;
}

// ---------------------------------------------------------------------------
// Zero LSTM state (h f32 + bf16, c f32).  32768 elements.
// ---------------------------------------------------------------------------
__global__ void init_state_kernel(float* __restrict__ h, bf16_t* __restrict__ hb,
                                  float* __restrict__ c)
{
    int i = blockIdx.x * blockDim.x + threadIdx.x;
    h[i]  = 0.f;
    c[i]  = 0.f;
    hb[i] = (bf16_t)0.f;
}

// ---------------------------------------------------------------------------
// Emissions init: em[b,t,k] = fc_b[k]
// ---------------------------------------------------------------------------
__global__ void em_init_kernel(float* __restrict__ em,
                               const float* __restrict__ fc_b, int total)
{
    int i = blockIdx.x * blockDim.x + threadIdx.x;
    if (i >= total) return;
    em[i] = fc_b[i % K_TAGS];
}

// ---------------------------------------------------------------------------
// Persistent one-direction LSTM:
//   grid = 64 blocks x 256 threads = 512 waves.
//   Phase A (per step): each wave owns one 16x16 tile of
//       gates = [xe_t ; h] * wcat^T + (b_ih+b_hh)   (M=64, N=2048, K=768)
//     24 x v_wmma_f32_16x16x32_bf16 per tile, fragments per ISA 7.12.2.
//   Phase B (per step): block b owns batch b: LSTM pointwise cell + masked
//     output into LDS + K=9 FC dot-products accumulated into emissions.
//   Device-wide barrier between phases and between steps.
// ---------------------------------------------------------------------------
__global__ __launch_bounds__(256) void lstm_dir_persistent(
    const bf16_t* __restrict__ xe,     // [B, T, E]
    const bf16_t* __restrict__ wcat,   // [4H, KC]
    const float*  __restrict__ b_ih,
    const float*  __restrict__ b_hh,
    const int*    __restrict__ x_len,
    const float*  __restrict__ fc_w,   // [K, 2H]
    float*  __restrict__ em,           // [B, T, K]
    float*  __restrict__ gates,        // [B, 4H]
    float*  __restrict__ c,            // [B, H]
    float*  __restrict__ h_f32,        // [B, H]
    bf16_t* __restrict__ h_bf16,       // [B, H]
    unsigned* __restrict__ bar,        // {cnt, gen}
    int t0, int tstep, int dirOff)     // fwd: (0,+1,0)  bwd: (T-1,-1,H)
{
    __shared__ float hm_s[H_DIM];

    const int  lane   = threadIdx.x & 31;
    const int  waveId = (blockIdx.x << 3) | (threadIdx.x >> 5);  // 0..511
    const int  tileM  = waveId & 3;
    const int  tileN  = waveId >> 2;
    const int  rowA   = tileM * 16 + (lane & 15);   // batch index
    const int  colB   = tileN * 16 + (lane & 15);   // gate/out-feature index
    const bool hiHalf = lane >= 16;

    const bf16_t* __restrict__ hrow = h_bf16 + (size_t)rowA * H_DIM;
    const bf16_t* __restrict__ wrow = wcat   + (size_t)colB * KC;
    const float bias = b_ih[colB] + b_hh[colB];

    const int b    = blockIdx.x;       // pointwise batch owner
    const int lenb = x_len[b];

    #pragma unroll 1
    for (int s = 0; s < T_DIM; ++s) {
        const int t = t0 + tstep * s;

        // ------------------ Phase A: WMMA GEMM (no divergence; EXEC all-1s)
        {
            const bf16_t* xrow = xe + ((size_t)rowA * T_DIM + t) * E_DIM;
            v8f acc = {};
            #pragma unroll
            for (int kc = 0; kc < KC; kc += 32) {
                const bf16_t* aptr = (kc < E_DIM) ? (xrow + kc) : (hrow + (kc - E_DIM));
                const bf16_t* bptr = wrow + kc;
                v16bf a, bb;
                // A fragment: 16-bit A 16x32 layout
                #pragma unroll
                for (int j = 0; j < 8; ++j) {
                    int ko = (j < 4) ? ((hiHalf ? 8 : 0) + 2 * j)
                                     : ((hiHalf ? 24 : 16) + 2 * (j - 4));
                    a[2 * j]     = aptr[ko];
                    a[2 * j + 1] = aptr[ko + 1];
                }
                // B fragment: 32x16 layout (K striped across lane halves)
                #pragma unroll
                for (int j = 0; j < 8; ++j) {
                    int ko = (j < 4) ? ((hiHalf ? 8 : 0) + 2 * j)
                                     : ((hiHalf ? 24 : 16) + 2 * (j - 4));
                    bb[2 * j]     = bptr[ko];
                    bb[2 * j + 1] = bptr[ko + 1];
                }
                acc = __builtin_amdgcn_wmma_f32_16x16x32_bf16(
                    false, a, false, bb, (short)0, acc, false, false);
            }
            // C layout: VGPR v -> M = v + (lane>=16 ? 8 : 0), N = lane&15
            #pragma unroll
            for (int v = 0; v < 8; ++v) {
                int m = tileM * 16 + v + (hiHalf ? 8 : 0);
                gates[(size_t)m * FOURH + colB] = acc[v] + bias;
            }
        }
        grid_sync(&bar[0], &bar[1]);   // gates visible to all blocks

        // ------------------ Phase B: pointwise cell + fused FC (block = batch)
        #pragma unroll
        for (int half = 0; half < 2; ++half) {
            const int j = threadIdx.x + half * 256;
            const float* g = gates + (size_t)b * FOURH;
            float i_ = 1.f / (1.f + __expf(-g[j]));
            float f_ = 1.f / (1.f + __expf(-g[H_DIM + j]));
            float gg = tanhf(g[2 * H_DIM + j]);
            float o_ = 1.f / (1.f + __expf(-g[3 * H_DIM + j]));

            const int idx = b * H_DIM + j;
            float c_old = c[idx];
            float h_old = h_f32[idx];
            float c_new = f_ * c_old + i_ * gg;
            float h_new = o_ * tanhf(c_new);

            bool v = t < lenb;
            float c_u = v ? c_new : c_old;
            float h_u = v ? h_new : h_old;
            c[idx]      = c_u;
            h_f32[idx]  = h_u;
            h_bf16[idx] = (bf16_t)h_u;
            hm_s[j]     = v ? h_new : 0.f;       // h * valid (per-step output)
        }
        __syncthreads();
        if (threadIdx.x < K_TAGS) {
            const float* w = fc_w + (size_t)threadIdx.x * (2 * H_DIM) + dirOff;
            float ssum = 0.f;
            for (int q = 0; q < H_DIM; ++q) ssum += hm_s[q] * w[q];
            em[((size_t)b * T_DIM + t) * K_TAGS + threadIdx.x] += ssum;
        }
        grid_sync(&bar[0], &bar[1]);   // h/c visible before next step's GEMM
    }
}

// ---------------------------------------------------------------------------
// CRF negative log-likelihood. One thread per batch element (K=9).
// ---------------------------------------------------------------------------
__global__ __launch_bounds__(64) void crf_loss_kernel(
    const float* __restrict__ em, const int* __restrict__ y,
    const int* __restrict__ x_len,
    const float* __restrict__ start, const float* __restrict__ trans,
    const float* __restrict__ end_, float* __restrict__ loss_out)
{
    __shared__ float red[B_DIM];
    const int b   = threadIdx.x;
    const int len = x_len[b];
    const int*   yb = y  + (size_t)b * T_DIM;
    const float* eb = em + (size_t)b * T_DIM * K_TAGS;

    // numerator
    float num = start[yb[0]] + eb[yb[0]];
    for (int t = 1; t < T_DIM; ++t) {
        if (t < len) {
            num += eb[t * K_TAGS + yb[t]];
            num += trans[yb[t - 1] * K_TAGS + yb[t]];
        }
    }
    num += end_[yb[len - 1]];

    // log-partition (forward algorithm)
    float sc[K_TAGS];
    for (int k = 0; k < K_TAGS; ++k) sc[k] = start[k] + eb[k];
    for (int t = 1; t < T_DIM; ++t) {
        if (t < len) {
            float nx[K_TAGS];
            for (int k2 = 0; k2 < K_TAGS; ++k2) {
                float m = -1e30f;
                for (int k = 0; k < K_TAGS; ++k)
                    m = fmaxf(m, sc[k] + trans[k * K_TAGS + k2]);
                float s = 0.f;
                for (int k = 0; k < K_TAGS; ++k)
                    s += __expf(sc[k] + trans[k * K_TAGS + k2] - m);
                nx[k2] = m + __logf(s) + eb[t * K_TAGS + k2];
            }
            for (int k2 = 0; k2 < K_TAGS; ++k2) sc[k2] = nx[k2];
        }
    }
    float m = -1e30f;
    for (int k = 0; k < K_TAGS; ++k) m = fmaxf(m, sc[k] + end_[k]);
    float s = 0.f;
    for (int k = 0; k < K_TAGS; ++k) s += __expf(sc[k] + end_[k] - m);
    float denom = m + __logf(s);

    red[b] = num - denom;
    __syncthreads();
    if (b == 0) {
        float tot = 0.f;
        for (int i = 0; i < B_DIM; ++i) tot += red[i];
        loss_out[0] = -tot;
    }
}

// ---------------------------------------------------------------------------
// Viterbi decode. One thread per batch element; backpointers in workspace.
// Tags written to d_out as floats.
// ---------------------------------------------------------------------------
__global__ __launch_bounds__(64) void viterbi_kernel(
    const float* __restrict__ em, const int* __restrict__ x_len,
    const float* __restrict__ start, const float* __restrict__ trans,
    const float* __restrict__ end_,
    int* __restrict__ hist, float* __restrict__ tags_out)
{
    const int b   = threadIdx.x;
    const int len = x_len[b];
    const float* eb = em   + (size_t)b * T_DIM * K_TAGS;
    int*         hb = hist + (size_t)b * T_DIM * K_TAGS;

    float sc[K_TAGS];
    for (int k = 0; k < K_TAGS; ++k) sc[k] = start[k] + eb[k];
    for (int t = 1; t < T_DIM; ++t) {
        bool m = t < len;
        for (int k2 = 0; k2 < K_TAGS; ++k2) {
            float best = -1e30f; int bi = 0;
            for (int k = 0; k < K_TAGS; ++k) {
                float v = sc[k] + trans[k * K_TAGS + k2];
                if (v > best) { best = v; bi = k; }
            }
            hb[t * K_TAGS + k2] = m ? bi : k2;
            if (m) sc[k2] = best + eb[t * K_TAGS + k2];
        }
    }
    int tag = 0;
    {
        float best = -1e30f;
        for (int k = 0; k < K_TAGS; ++k) {
            float v = sc[k] + end_[k];
            if (v > best) { best = v; tag = k; }
        }
    }
    tags_out[(size_t)b * T_DIM + (T_DIM - 1)] = (float)tag;
    for (int t = T_DIM - 1; t >= 1; --t) {
        tag = hb[t * K_TAGS + tag];
        tags_out[(size_t)b * T_DIM + (t - 1)] = (float)tag;
    }
}

// ---------------------------------------------------------------------------
// Host launch (11 graph nodes total)
// ---------------------------------------------------------------------------
extern "C" void kernel_launch(void* const* d_in, const int* in_sizes, int n_in,
                              void* d_out, int out_size, void* d_ws, size_t ws_size,
                              hipStream_t stream)
{
    (void)in_sizes; (void)n_in; (void)out_size; (void)ws_size;

    const int*   x       = (const int*)  d_in[0];
    const int*   x_len   = (const int*)  d_in[1];
    const int*   y       = (const int*)  d_in[2];
    const float* embed   = (const float*)d_in[3];
    const float* w_ih_f  = (const float*)d_in[4];
    const float* w_hh_f  = (const float*)d_in[5];
    const float* b_ih_f  = (const float*)d_in[6];
    const float* b_hh_f  = (const float*)d_in[7];
    const float* w_ih_b  = (const float*)d_in[8];
    const float* w_hh_b  = (const float*)d_in[9];
    const float* b_ih_b  = (const float*)d_in[10];
    const float* b_hh_b  = (const float*)d_in[11];
    const float* fc_w    = (const float*)d_in[12];
    const float* fc_b    = (const float*)d_in[13];
    const float* crf_start = (const float*)d_in[14];
    const float* crf_trans = (const float*)d_in[15];
    const float* crf_end   = (const float*)d_in[16];
    float* out = (float*)d_out;

    // ---- workspace carving (256B aligned) ----
    char* wsp = (char*)d_ws;
    auto carve = [&](size_t bytes) -> void* {
        void* p = (void*)wsp;
        wsp += (bytes + 255) & ~(size_t)255;
        return p;
    };
    bf16_t*   xe      = (bf16_t*)  carve((size_t)B_DIM * T_DIM * E_DIM * 2); // 16.8 MB
    bf16_t*   wcat_f  = (bf16_t*)  carve((size_t)FOURH * KC * 2);            // 3.1 MB
    bf16_t*   wcat_b  = (bf16_t*)  carve((size_t)FOURH * KC * 2);            // 3.1 MB
    float*    gates   = (float*)   carve((size_t)B_DIM * FOURH * 4);         // 512 KB
    float*    h_f32   = (float*)   carve((size_t)B_DIM * H_DIM * 4);
    bf16_t*   h_bf16  = (bf16_t*)  carve((size_t)B_DIM * H_DIM * 2);
    float*    c_st    = (float*)   carve((size_t)B_DIM * H_DIM * 4);
    float*    em      = (float*)   carve((size_t)B_DIM * T_DIM * K_TAGS * 4); // 1.2 MB
    int*      hist    = (int*)     carve((size_t)B_DIM * T_DIM * K_TAGS * 4); // 1.2 MB
    unsigned* bar     = (unsigned*)carve(256);  // bar[0..1]=fwd, bar[2..3]=bwd

    // ---- setup ----
    barrier_init_kernel<<<1, 32, 0, stream>>>(bar);
    {
        int total = B_DIM * T_DIM * E_DIM;
        embed_bf16_kernel<<<(total + 255) / 256, 256, 0, stream>>>(x, embed, xe, total);
    }
    {
        int total = FOURH * KC;
        wcat_kernel<<<(total + 255) / 256, 256, 0, stream>>>(w_ih_f, w_hh_f, wcat_f, total);
        wcat_kernel<<<(total + 255) / 256, 256, 0, stream>>>(w_ih_b, w_hh_b, wcat_b, total);
    }
    {
        int total = B_DIM * T_DIM * K_TAGS;
        em_init_kernel<<<(total + 255) / 256, 256, 0, stream>>>(em, fc_b, total);
    }

    // ---- forward-direction LSTM (one persistent kernel, 512 steps) ----
    init_state_kernel<<<(B_DIM * H_DIM) / 256, 256, 0, stream>>>(h_f32, h_bf16, c_st);
    lstm_dir_persistent<<<NBLK, 256, 0, stream>>>(
        xe, wcat_f, b_ih_f, b_hh_f, x_len, fc_w, em,
        gates, c_st, h_f32, h_bf16, &bar[0], /*t0=*/0, /*tstep=*/+1, /*dirOff=*/0);

    // ---- backward-direction LSTM ----
    init_state_kernel<<<(B_DIM * H_DIM) / 256, 256, 0, stream>>>(h_f32, h_bf16, c_st);
    lstm_dir_persistent<<<NBLK, 256, 0, stream>>>(
        xe, wcat_b, b_ih_b, b_hh_b, x_len, fc_w, em,
        gates, c_st, h_f32, h_bf16, &bar[2], /*t0=*/T_DIM - 1, /*tstep=*/-1, /*dirOff=*/H_DIM);

    // ---- CRF loss + Viterbi decode ----
    crf_loss_kernel<<<1, B_DIM, 0, stream>>>(em, y, x_len, crf_start, crf_trans,
                                             crf_end, out + (size_t)B_DIM * T_DIM);
    viterbi_kernel<<<1, B_DIM, 0, stream>>>(em, x_len, crf_start, crf_trans,
                                            crf_end, hist, out);
}
